// TraditionalQNN_33672543600857
// MI455X (gfx1250) — compile-verified, matrix operand order checked by reading, not run on hardware
//
#include <hip/hip_runtime.h>

typedef __attribute__((ext_vector_type(16))) _Float16 v16h;
typedef __attribute__((ext_vector_type(8)))  float    v8f;

#define NROWS (256 * 512)   // B*T = 131072 rows
// Workspace layout (floats)
#define WS_A    0           // 128: A_comb[4][32] = w_cip @ w_in
#define WS_C4   128         // 4  : c4 = w_cip @ b_in
#define WS_M    132         // 40 : M10[10][4] = w_head @ w_coe
#define WS_CW   172         // 8  : cos(q_weights/2)
#define WS_SW   180         // 8  : sin(q_weights/2)
#define WS_TOT  188

// ---------------- prep: fold the big weight matrices into tiny tables ----------------
__global__ void qnn_prep(const float* __restrict__ w_in, const float* __restrict__ b_in,
                         const float* __restrict__ w_cip, const float* __restrict__ q_weights,
                         const float* __restrict__ w_coe, const float* __restrict__ w_head,
                         float* __restrict__ ws) {
    int t = threadIdx.x;              // 128 threads
    {   // A_comb[a][k] = sum_e w_cip[a][e] * w_in[e][k]   (4x32)
        int a = t >> 5, k = t & 31;
        float acc = 0.f;
        for (int e = 0; e < 128; ++e) acc += w_cip[a * 128 + e] * w_in[e * 32 + k];
        ws[WS_A + t] = acc;
    }
    if (t < 4) {                      // c4[a] = sum_e w_cip[a][e] * b_in[e]
        float acc = 0.f;
        for (int e = 0; e < 128; ++e) acc += w_cip[t * 128 + e] * b_in[e];
        ws[WS_C4 + t] = acc;
    }
    if (t < 40) {                     // M10[o][a] = sum_e w_head[o][e] * w_coe[e][a]
        int o = t >> 2, a = t & 3;
        float acc = 0.f;
        for (int e = 0; e < 128; ++e) acc += w_head[o * 128 + e] * w_coe[e * 4 + a];
        ws[WS_M + t] = acc;
    }
    if (t < 8) {                      // half-angle cos/sin of entangler weights
        float h = q_weights[t] * 0.5f;
        float s, c;
        __sincosf(h, &s, &c);
        ws[WS_CW + t] = c;
        ws[WS_SW + t] = s;
    }
}

// ---------------- 4-qubit gate helpers (fully unrolled, register resident) ----------------
// Wire w acts on bit of weight 1<<(3-w) in the flat 16-state index.
template <int ST>
__device__ __forceinline__ void rx_apply(float* re, float* im, float c, float s) {
#pragma unroll
    for (int i = 0; i < 16; ++i) {
        if (i & ST) continue;
        const int j = i | ST;
        float r0 = re[i], i0 = im[i], r1 = re[j], i1 = im[j];
        re[i] = c * r0 + s * i1;      // new0 = c*a0 - i*s*a1
        im[i] = c * i0 - s * r1;
        re[j] = c * r1 + s * i0;      // new1 = c*a1 - i*s*a0
        im[j] = c * i1 - s * r0;
    }
}

template <int SC, int ST>
__device__ __forceinline__ void cnot_apply(float* re, float* im) {
#pragma unroll
    for (int i = 0; i < 16; ++i) {
        if ((i & SC) && !(i & ST)) {
            const int j = i | ST;
            float tr = re[i], ti = im[i];
            re[i] = re[j]; im[i] = im[j];
            re[j] = tr;    im[j] = ti;
        }
    }
}

// ---------------- main: one row per lane, WMMA head, LDS-staged coalesced output ----------------
__launch_bounds__(256)
__global__ void qnn_main(const float* __restrict__ inputs, const float* __restrict__ ws,
                         float* __restrict__ out) {
    __shared__ float smem[WS_TOT];                 // folded weight tables
    __shared__ alignas(16) float sout[256 * 10];   // staged block output (2560 floats)

    const int tid = threadIdx.x;
    if (tid < WS_TOT) smem[tid] = ws[tid];
    __syncthreads();

    const int lane    = tid & 31;
    const int wave_id = tid >> 5;
    const int row     = blockIdx.x * 256 + tid;    // exact division: always in range

    // ---- stage 1: 32 -> 4 angles (per-lane dot products, weights broadcast from LDS) ----
    float x[32];
    const float4* xp = (const float4*)(inputs + (size_t)row * 32);
#pragma unroll
    for (int i = 0; i < 8; ++i) {
        float4 v = xp[i];
        x[4 * i + 0] = v.x; x[4 * i + 1] = v.y; x[4 * i + 2] = v.z; x[4 * i + 3] = v.w;
    }
    float ang[4];
#pragma unroll
    for (int a = 0; a < 4; ++a) {
        float acc = smem[WS_C4 + a];
#pragma unroll
        for (int k = 0; k < 32; ++k) acc += smem[WS_A + a * 32 + k] * x[k];
        ang[a] = acc;
    }

    // ---- stage 2: 4-qubit statevector simulation ----
    float re[16], im[16];
#pragma unroll
    for (int i = 0; i < 16; ++i) { re[i] = 0.f; im[i] = 0.f; }
    re[0] = 1.f;

    // AngleEmbedding: RX(ang[w]) on wire w
    {
        float s, c;
        __sincosf(ang[0] * 0.5f, &s, &c); rx_apply<8>(re, im, c, s);
        __sincosf(ang[1] * 0.5f, &s, &c); rx_apply<4>(re, im, c, s);
        __sincosf(ang[2] * 0.5f, &s, &c); rx_apply<2>(re, im, c, s);
        __sincosf(ang[3] * 0.5f, &s, &c); rx_apply<1>(re, im, c, s);
    }
    // BasicEntanglerLayers: RX(q_weights[l][w]) then CNOT ring
#pragma unroll
    for (int l = 0; l < 2; ++l) {
        rx_apply<8>(re, im, smem[WS_CW + l * 4 + 0], smem[WS_SW + l * 4 + 0]);
        rx_apply<4>(re, im, smem[WS_CW + l * 4 + 1], smem[WS_SW + l * 4 + 1]);
        rx_apply<2>(re, im, smem[WS_CW + l * 4 + 2], smem[WS_SW + l * 4 + 2]);
        rx_apply<1>(re, im, smem[WS_CW + l * 4 + 3], smem[WS_SW + l * 4 + 3]);
        cnot_apply<8, 4>(re, im);   // (0,1)
        cnot_apply<4, 2>(re, im);   // (1,2)
        cnot_apply<2, 1>(re, im);   // (2,3)
        cnot_apply<1, 8>(re, im);   // (3,0)
    }

    // PauliZ expectation per wire
    float p[16];
#pragma unroll
    for (int i = 0; i < 16; ++i) p[i] = re[i] * re[i] + im[i] * im[i];
    float q0 = 0.f, q1 = 0.f, q2 = 0.f, q3 = 0.f;
#pragma unroll
    for (int i = 0; i < 16; ++i) {
        q0 += (i & 8) ? -p[i] : p[i];
        q1 += (i & 4) ? -p[i] : p[i];
        q2 += (i & 2) ? -p[i] : p[i];
        q3 += (i & 1) ? -p[i] : p[i];
    }

    // ---- stage 3: 4 -> 10 head via v_wmma_f32_16x16x32_f16 ----
    // B = M10^T padded to 32x16 f16: lanes 0-15 hold K=0..7 and 16..23 of column N=lane.
    v16h bmat = {};
    {
        const int n = lane & 15;
        if (lane < 16 && n < 10) {
#pragma unroll
            for (int k = 0; k < 4; ++k) bmat[k] = (_Float16)smem[WS_M + n * 4 + k];
        }
    }

#pragma unroll
    for (int g = 0; g < 2; ++g) {
        // A = Q[g*16 .. g*16+15][0..3] padded to 16x32 f16.
        // f16 A layout: lanes 0-15 carry K=0..7 (elements 0-7) for M=lane -> q0..q3 in elems 0-3.
        const int src = (lane & 15) + g * 16;        // owner lane of row M = lane&15 in group g
        float qa0 = __shfl(q0, src, 32);
        float qa1 = __shfl(q1, src, 32);
        float qa2 = __shfl(q2, src, 32);
        float qa3 = __shfl(q3, src, 32);
        v16h amat = {};
        if (lane < 16) {
            amat[0] = (_Float16)qa0; amat[1] = (_Float16)qa1;
            amat[2] = (_Float16)qa2; amat[3] = (_Float16)qa3;
        }
        v8f cz = {};
        v8f d = __builtin_amdgcn_wmma_f32_16x16x32_f16(
            /*neg_a=*/false, amat, /*neg_b=*/false, bmat,
            /*c_mod=*/(short)0, cz, /*reuse_a=*/false, /*reuse_b=*/false);

        // D layout: lanes 0-15 -> (M=j, N=lane); lanes 16-31 -> (M=8+j, N=lane-16).
        // Scatter into the block-local LDS staging buffer (cheap LDS scatter,
        // bank-conflict-free: 10 consecutive words per half-wave, halves offset
        // by 80 words == 16 mod 64 banks).
        const int ch     = lane & 15;
        const int rlocal = wave_id * 32 + g * 16 + ((lane < 16) ? 0 : 8);
        if (ch < 10) {
#pragma unroll
            for (int j = 0; j < 8; ++j)
                sout[(rlocal + j) * 10 + ch] = d[j];
        }
    }

    // ---- coalesced block write-out: 2560 contiguous floats = 640 float4 ----
    __syncthreads();
    const float4* s4 = (const float4*)sout;
    float4* o4 = (float4*)(out + (size_t)blockIdx.x * 2560);
    o4[tid]       = s4[tid];
    o4[tid + 256] = s4[tid + 256];
    if (tid < 128)
        o4[tid + 512] = s4[tid + 512];
}

extern "C" void kernel_launch(void* const* d_in, const int* in_sizes, int n_in,
                              void* d_out, int out_size, void* d_ws, size_t ws_size,
                              hipStream_t stream) {
    (void)in_sizes; (void)n_in; (void)out_size; (void)ws_size;
    const float* inputs   = (const float*)d_in[0];
    const float* w_in     = (const float*)d_in[1];
    const float* b_in     = (const float*)d_in[2];
    const float* w_cip    = (const float*)d_in[3];
    const float* q_wts    = (const float*)d_in[4];
    const float* w_coe    = (const float*)d_in[5];
    const float* w_head   = (const float*)d_in[6];
    float* out = (float*)d_out;
    float* ws  = (float*)d_ws;

    qnn_prep<<<1, 128, 0, stream>>>(w_in, b_in, w_cip, q_wts, w_coe, w_head, ws);
    qnn_main<<<NROWS / 256, 256, 0, stream>>>(inputs, ws, out);
}
